// DNCEncoder_73126113181804
// MI455X (gfx1250) — compile-verified, matrix operand order checked by reading, not run on hardware
//
#include <hip/hip_runtime.h>
#include <hip/hip_bf16.h>

// ---------------------------------------------------------------------------
// DNC encoder forward for MI455X (gfx1250, wave32, WMMA).
// B=16,S=128,H=512,V=50000,N=5,R=2,Wd=512,XI=2573
// Single 1024-thread workgroup runs the sequential recurrence with all state
// in LDS; three bf16 WMMA GEMMs per step (gates 2048x2048, xi 512x2576(pad),
// out 1536x512); weights pre-packed to per-lane WMMA B-fragment layout.
// ---------------------------------------------------------------------------

typedef __bf16 bf16;
typedef __bf16 v16bf __attribute__((ext_vector_type(16)));
typedef float  v8f   __attribute__((ext_vector_type(8)));

struct Frag32B { uint4 lo, hi; };   // 32 bytes = 16 bf16 = one WMMA operand per lane

static constexpr int Bk = 16, Sk = 128, Hk = 512;
static constexpr int XI_PAD = 2576;                 // 2573 padded to 161*16
// workspace byte offsets
static constexpr size_t WG_BYTES = (size_t)128 * 64 * 32 * 16 * 2;   // 8 MB
static constexpr size_t WI_BYTES = (size_t)161 * 16 * 32 * 16 * 2;   // 2.52 MB
static constexpr size_t WO_BYTES = (size_t)32  * 48 * 32 * 16 * 2;   // 1.5 MB
static constexpr size_t XI_OFF   = WG_BYTES + WI_BYTES + WO_BYTES;
// LDS layout
static constexpr int SMEM_BYTES = 65536 /*act bf16 16x2048*/ +
    4 * (8192 /*c*/ + 8192 /*h*/ + 40960 /*M*/ + 800 /*u,p,L,wr,ww*/ + 272 /*dots*/);

__device__ __forceinline__ unsigned short f2bf(float f) {
  unsigned int u = __builtin_bit_cast(unsigned int, f);
  unsigned int r = u + 0x7FFFu + ((u >> 16) & 1u);
  if ((u & 0x7F800000u) == 0x7F800000u) r = u;       // inf/nan passthrough
  return (unsigned short)(r >> 16);
}
__device__ __forceinline__ float sigm(float x) { return 1.f / (1.f + __expf(-x)); }
__device__ __forceinline__ float oneplusf(float x) {
  return 1.f + ((x > 20.f) ? x : log1pf(__expf(x)));
}
__device__ __forceinline__ float wave_sum(float x) {
  for (int o = 16; o > 0; o >>= 1) x += __shfl_xor(x, o, 32);
  return x;
}

__device__ __forceinline__ v8f wmma_bf16(v16bf a, v16bf b, v8f c) {
  return __builtin_amdgcn_wmma_f32_16x16x32_bf16(false, a, false, b, (short)0, c,
                                                 false, false);
}

// B fragment: packed global, frag-major [frag][lane][16] bf16 (1 KB / frag)
__device__ __forceinline__ v16bf load_bfrag(const unsigned short* p, int frag, int lane) {
  const uint4* q = (const uint4*)(p + ((size_t)frag * 32 + lane) * 16);
  Frag32B f; f.lo = q[0]; f.hi = q[1];
  return __builtin_bit_cast(v16bf, f);
}
// A fragment from LDS row-major act[16][2048] bf16 at column offset colofs
//   lane<16 : M=lane,    K = kt*32 + {0..7, 16..23}
//   lane>=16: M=lane-16, K = kt*32 + {8..15, 24..31}
__device__ __forceinline__ v16bf load_afrag(const unsigned short* act, int lane,
                                            int kt, int colofs) {
  int m  = lane & 15;
  int kb = kt * 32 + ((lane >> 4) << 3);
  const unsigned short* row = act + m * 2048 + colofs + kb;
  Frag32B f;
  f.lo = *(const uint4*)(row);
  f.hi = *(const uint4*)(row + 16);
  return __builtin_bit_cast(v16bf, f);
}

// ---------------- weight packing kernels (fp32 -> bf16 WMMA fragments) ------
// element e of a lane:  k_local = ((e>>3)<<4) + ((lane>>4)<<3) + (e&7), n_local = lane&15

__global__ void pack_gates(const float* __restrict__ Wih, const float* __restrict__ Whh,
                           unsigned short* __restrict__ out) {
  int t = blockIdx.x * blockDim.x + threadIdx.x;
  int lane = t & 31, frag = t >> 5;
  if (frag >= 128 * 64) return;
  int nt = frag >> 6, kt = frag & 63;
  int n = nt * 16 + (lane & 15);
  int kbase = kt * 32 + ((lane >> 4) << 3);
  unsigned short* dst = out + ((size_t)frag * 32 + lane) * 16;
  for (int e = 0; e < 16; ++e) {
    int k = kbase + ((e >> 3) << 4) + (e & 7);
    float v = (k < 1536) ? Wih[(size_t)n * 1536 + k] : Whh[(size_t)n * 512 + (k - 1536)];
    dst[e] = f2bf(v);
  }
}

__global__ void pack_int(const float* __restrict__ Wint, unsigned short* __restrict__ out) {
  int t = blockIdx.x * blockDim.x + threadIdx.x;
  int lane = t & 31, frag = t >> 5;
  if (frag >= 161 * 16) return;
  int nt = frag / 16, kt = frag % 16;
  int n = nt * 16 + (lane & 15);
  int kbase = kt * 32 + ((lane >> 4) << 3);
  unsigned short* dst = out + ((size_t)frag * 32 + lane) * 16;
  for (int e = 0; e < 16; ++e) {
    int k = kbase + ((e >> 3) << 4) + (e & 7);
    float v = (n < 2573) ? Wint[(size_t)k * 2573 + n] : 0.f;
    dst[e] = f2bf(v);
  }
}

// W_out rows permuted so A = act[:,512:2048] = [r(1024) | h(512)] is contiguous
__global__ void pack_out(const float* __restrict__ Wout, unsigned short* __restrict__ out) {
  int t = blockIdx.x * blockDim.x + threadIdx.x;
  int lane = t & 31, frag = t >> 5;
  if (frag >= 32 * 48) return;
  int nt = frag / 48, kt = frag % 48;
  int n = nt * 16 + (lane & 15);
  int kbase = kt * 32 + ((lane >> 4) << 3);
  unsigned short* dst = out + ((size_t)frag * 32 + lane) * 16;
  for (int e = 0; e < 16; ++e) {
    int j = kbase + ((e >> 3) << 4) + (e & 7);
    int row = (j < 1024) ? (512 + j) : (j - 1024);   // r rows then h rows
    dst[e] = f2bf(Wout[(size_t)row * 512 + n]);
  }
}

// ---------------- main recurrent kernel -------------------------------------
__global__ __launch_bounds__(1024) void dnc_main(
    const int*  __restrict__ source, const float* __restrict__ emb,
    const float* __restrict__ b_lstm, const float* __restrict__ b_int,
    const float* __restrict__ b_out,
    const unsigned short* __restrict__ wg, const unsigned short* __restrict__ wi,
    const unsigned short* __restrict__ wo,
    float* __restrict__ xiW, float* __restrict__ out) {
  extern __shared__ unsigned char smem[];
  unsigned short* actS = (unsigned short*)smem;       // [16][2048] bf16: [x|r0|r1|h]
  float* cS   = (float*)(smem + 65536);               // [16][512]
  float* hS   = cS + 8192;                            // [16][512]
  float* MS   = hS + 8192;                            // [16][5][512]
  float* uS   = MS + 40960;                           // [16][5]
  float* pS   = uS + 80;                              // [16][5]
  float* LSm  = pS + 80;                              // [16][25]
  float* wrS  = LSm + 400;                            // [16][2][5]
  float* wwS  = wrS + 160;                            // [16][5]
  float* dotS = wwS + 80;                             // [16][17]

  const int tid  = threadIdx.x;
  const int lane = tid & 31;
  const int wave = tid >> 5;                          // 0..31
  const int bb   = wave >> 1;                         // batch for DNC phases
  const int hw   = wave & 1;                          // sub-wave within batch

  // ---- init state ----
  for (int i = tid; i < 8192; i += 1024) { cS[i] = 0.f; hS[i] = 0.f; }
  for (int i = tid; i < 40960; i += 1024) MS[i] = 0.f;
  for (int i = tid; i < 800; i += 1024) uS[i] = 0.f;  // u,p,L,wr,ww contiguous
  for (int i = tid; i < 16 * 1536; i += 1024) {       // zero r|h part of act
    int m = i / 1536, j = i - m * 1536;
    actS[m * 2048 + 512 + j] = 0;
  }
  __syncthreads();

  for (int s = 0; s < Sk; ++s) {
    // ---- embedding lookup -> act[:,0:512] ----
    for (int i = tid; i < 8192; i += 1024) {
      int m = i >> 9, w = i & 511;
      int tok = source[m * Sk + s];
      actS[m * 2048 + w] = f2bf(emb[(size_t)tok * Hk + w]);
    }
    __syncthreads();

    // ---- gates GEMM (K=2048) fused with LSTM pointwise ----
    {
      int ct = wave;                                  // 16-col slice of each gate
      v8f ai, af, ag, ao;
      for (int v = 0; v < 8; ++v) { ai[v] = 0.f; af[v] = 0.f; ag[v] = 0.f; ao[v] = 0.f; }
      for (int kt = 0; kt < 64; ++kt) {
        v16bf a  = load_afrag(actS, lane, kt, 0);
        v16bf b0 = load_bfrag(wg, (ct)      * 64 + kt, lane);
        v16bf b1 = load_bfrag(wg, (ct + 32) * 64 + kt, lane);
        v16bf b2 = load_bfrag(wg, (ct + 64) * 64 + kt, lane);
        v16bf b3 = load_bfrag(wg, (ct + 96) * 64 + kt, lane);
        ai = wmma_bf16(a, b0, ai);
        af = wmma_bf16(a, b1, af);
        ag = wmma_bf16(a, b2, ag);
        ao = wmma_bf16(a, b3, ao);
      }
      int col = ct * 16 + (lane & 15);
      int mb  = (lane >> 4) << 3;
      float bi_ = b_lstm[col], bf_ = b_lstm[512 + col];
      float bg_ = b_lstm[1024 + col], bo_ = b_lstm[1536 + col];
      for (int v = 0; v < 8; ++v) {
        int m = mb + v;
        float cc = sigm(af[v] + bf_) * cS[m * 512 + col] +
                   sigm(ai[v] + bi_) * tanhf(ag[v] + bg_);
        float hh = sigm(ao[v] + bo_) * tanhf(cc);
        cS[m * 512 + col] = cc;
        hS[m * 512 + col] = hh;
        actS[m * 2048 + 1536 + col] = f2bf(hh);
      }
    }
    __syncthreads();

    // ---- xi GEMM: h @ W_int (K=512, N=2576 padded) -> global ws ----
    for (int nt = wave; nt < 161; nt += 32) {
      v8f acc; for (int v = 0; v < 8; ++v) acc[v] = 0.f;
      for (int kt = 0; kt < 16; ++kt) {
        v16bf a = load_afrag(actS, lane, kt, 1536);
        v16bf b = load_bfrag(wi, nt * 16 + kt, lane);
        acc = wmma_bf16(a, b, acc);
      }
      int col = nt * 16 + (lane & 15);
      int mb  = (lane >> 4) << 3;
      float bias = (col < 2573) ? b_int[col] : 0.f;
      for (int v = 0; v < 8; ++v)
        xiW[(size_t)(mb + v) * XI_PAD + col] = acc[v] + bias;
    }
    __threadfence();
    __syncthreads();

    // ---- DNC phase A: dots over M_old / wk  (11 jobs, 2 waves per batch) ----
    {
      const float* xb = xiW + (size_t)bb * XI_PAD;
      for (int j = hw; j < 11; j += 2) {
        const float *x, *y;
        if (j < 5)       { x = y = MS + bb * 2560 + j * 512; }
        else if (j < 10) { x = xb + 1026; y = MS + bb * 2560 + (j - 5) * 512; }
        else             { x = y = xb + 1026; }
        float p = 0.f;
        for (int i = lane; i < 512; i += 32) p += x[i] * y[i];
        p = wave_sum(p);
        if (lane == 0) dotS[bb * 17 + j] = p;
      }
    }
    __syncthreads();

    // ---- DNC serial: usage, allocation(sort), write content, ww ----
    if (lane == 0 && hw == 0) {
      const float* xb = xiW + (size_t)bb * XI_PAD;
      float fg0 = sigm(xb[2563]), fg1 = sigm(xb[2564]);
      float uo[5];
      for (int n = 0; n < 5; ++n) {
        float ps = (1.f - fg0 * wrS[bb * 10 + n]) * (1.f - fg1 * wrS[bb * 10 + 5 + n]);
        float uu = uS[bb * 5 + n], w = wwS[bb * 5 + n];
        uo[n] = (uu + w - uu * w) * ps;
        uS[bb * 5 + n] = uo[n];
      }
      int idx[5] = {0, 1, 2, 3, 4};                   // stable insertion sort (argsort)
      for (int a1 = 1; a1 < 5; ++a1) {
        int key = idx[a1], a2 = a1 - 1;
        while (a2 >= 0 && uo[idx[a2]] > uo[key]) { idx[a2 + 1] = idx[a2]; --a2; }
        idx[a2 + 1] = key;
      }
      float alloc[5], excl = 1.f;
      for (int t = 0; t < 5; ++t) {
        alloc[idx[t]] = (1.f - uo[idx[t]]) * excl;
        excl *= uo[idx[t]];
      }
      float wbv = oneplusf(xb[1538]);
      float nk = sqrtf(dotS[bb * 17 + 10]) + 1e-6f;
      float sim[5], mx = -1e30f;
      for (int n = 0; n < 5; ++n) {
        sim[n] = wbv * dotS[bb * 17 + 5 + n] / (nk * (sqrtf(dotS[bb * 17 + n]) + 1e-6f));
        mx = fmaxf(mx, sim[n]);
      }
      float se = 0.f;
      for (int n = 0; n < 5; ++n) { sim[n] = __expf(sim[n] - mx); se += sim[n]; }
      float ga = sigm(xb[2565]), gwv = sigm(xb[2566]);
      for (int n = 0; n < 5; ++n)
        wwS[bb * 5 + n] = gwv * (ga * alloc[n] + (1.f - ga) * (sim[n] / se));
    }
    __syncthreads();

    // ---- memory write: M = M*(1 - ww*e) + ww*v  (5x512 per batch, 64 thr) ----
    {
      const float* xb = xiW + (size_t)bb * XI_PAD;
      for (int i2 = hw * 32 + lane; i2 < 2560; i2 += 64) {
        int n = i2 >> 9, w = i2 & 511;
        float e = sigm(xb[1539 + w]);
        float v = xb[2051 + w];
        float wn = wwS[bb * 5 + n];
        MS[bb * 2560 + i2] = MS[bb * 2560 + i2] * (1.f - wn * e) + wn * v;
      }
    }
    __syncthreads();

    // ---- DNC phase B dots: M_new norms, rk.M, rk norms (17 jobs) ----
    {
      const float* xb = xiW + (size_t)bb * XI_PAD;
      for (int j = hw; j < 17; j += 2) {
        const float *x, *y;
        if (j < 5)       { x = y = MS + bb * 2560 + j * 512; }
        else if (j < 15) { int r = (j - 5) / 5, n = (j - 5) % 5;
                           x = xb + r * 512; y = MS + bb * 2560 + n * 512; }
        else             { x = y = xb + (j - 15) * 512; }
        float p = 0.f;
        for (int i = lane; i < 512; i += 32) p += x[i] * y[i];
        p = wave_sum(p);
        if (lane == 0) dotS[bb * 17 + j] = p;
      }
    }
    __syncthreads();

    // ---- DNC serial: link matrix, precedence, read weightings ----
    if (lane == 0 && hw == 0) {
      const float* xb = xiW + (size_t)bb * XI_PAD;
      float wwl[5], pol[5], Lo[25], wro[10], wwsum = 0.f;
      for (int n = 0; n < 5; ++n) {
        wwl[n] = wwS[bb * 5 + n]; wwsum += wwl[n]; pol[n] = pS[bb * 5 + n];
      }
      for (int k = 0; k < 25; ++k) Lo[k] = LSm[bb * 25 + k];
      for (int k = 0; k < 10; ++k) wro[k] = wrS[bb * 10 + k];
      for (int i = 0; i < 5; ++i)
        for (int j = 0; j < 5; ++j)
          LSm[bb * 25 + i * 5 + j] = (i == j) ? 0.f
              : (1.f - wwl[i] - wwl[j]) * Lo[i * 5 + j] + wwl[i] * pol[j];
      for (int n = 0; n < 5; ++n) pS[bb * 5 + n] = (1.f - wwsum) * pol[n] + wwl[n];
      for (int r = 0; r < 2; ++r) {
        float rbv = oneplusf(xb[1024 + r]);
        float nrk = sqrtf(dotS[bb * 17 + 15 + r]) + 1e-6f;
        float sim[5], mx = -1e30f;
        for (int n = 0; n < 5; ++n) {
          sim[n] = rbv * dotS[bb * 17 + 5 + r * 5 + n] /
                   (nrk * (sqrtf(dotS[bb * 17 + n]) + 1e-6f));
          mx = fmaxf(mx, sim[n]);
        }
        float se = 0.f;
        for (int n = 0; n < 5; ++n) { sim[n] = __expf(sim[n] - mx); se += sim[n]; }
        float p0 = xb[2567 + 3 * r], p1 = xb[2568 + 3 * r], p2 = xb[2569 + 3 * r];
        float pm = fmaxf(p0, fmaxf(p1, p2));
        float e0 = __expf(p0 - pm), e1 = __expf(p1 - pm), e2 = __expf(p2 - pm);
        float es = e0 + e1 + e2;
        for (int i = 0; i < 5; ++i) {
          float fwv = 0.f, bwv = 0.f;
          for (int j = 0; j < 5; ++j) {
            fwv += LSm[bb * 25 + i * 5 + j] * wro[r * 5 + j];
            bwv += LSm[bb * 25 + j * 5 + i] * wro[r * 5 + j];
          }
          wrS[bb * 10 + r * 5 + i] =
              (e0 * bwv + e1 * (sim[i] / se) + e2 * fwv) / es;
        }
      }
    }
    __syncthreads();

    // ---- read vectors r = wr . M -> act[:,512:1536] (wave hw = head hw) ----
    for (int i = lane; i < 512; i += 32) {
      float acc = 0.f;
      for (int n = 0; n < 5; ++n)
        acc += wrS[bb * 10 + hw * 5 + n] * MS[bb * 2560 + n * 512 + i];
      actS[bb * 2048 + 512 + hw * 512 + i] = f2bf(acc);
    }
    __syncthreads();

    // ---- out GEMM: [h|r] @ W_out (K=1536 via permuted pack, N=512) ----
    {
      int nt = wave;
      v8f acc; for (int v = 0; v < 8; ++v) acc[v] = 0.f;
      for (int kt = 0; kt < 48; ++kt) {
        v16bf a = load_afrag(actS, lane, kt, 512);
        v16bf b = load_bfrag(wo, nt * 48 + kt, lane);
        acc = wmma_bf16(a, b, acc);
      }
      int col = nt * 16 + (lane & 15);
      int mb  = (lane >> 4) << 3;
      float bias = b_out[col];
      for (int v = 0; v < 8; ++v)
        out[((size_t)(mb + v) * Sk + s) * Hk + col] = acc[v] + bias;
    }
    __syncthreads();
  }

  // ---- final (h, c) appended after outputs [B,S,H] ----
  for (int i = tid; i < 8192; i += 1024) {
    out[(size_t)Bk * Sk * Hk + i] = hS[i];
    out[(size_t)Bk * Sk * Hk + 8192 + i] = cS[i];
  }
}

// ---------------------------------------------------------------------------
extern "C" void kernel_launch(void* const* d_in, const int* in_sizes, int n_in,
                              void* d_out, int out_size, void* d_ws, size_t ws_size,
                              hipStream_t stream) {
  const int*   source  = (const int*)d_in[0];
  // d_in[1] = source_lengths (unused by reference computation)
  const float* emb     = (const float*)d_in[2];
  const float* W_ih    = (const float*)d_in[3];
  const float* W_hh    = (const float*)d_in[4];
  const float* b_lstm  = (const float*)d_in[5];
  const float* W_int   = (const float*)d_in[6];
  const float* b_int   = (const float*)d_in[7];
  const float* W_out   = (const float*)d_in[8];
  const float* b_out   = (const float*)d_in[9];

  unsigned char* ws = (unsigned char*)d_ws;
  unsigned short* wg  = (unsigned short*)ws;
  unsigned short* wi  = (unsigned short*)(ws + WG_BYTES);
  unsigned short* wo  = (unsigned short*)(ws + WG_BYTES + WI_BYTES);
  float*          xiW = (float*)(ws + XI_OFF);
  float*          outp = (float*)d_out;

  pack_gates<<<(128 * 64 * 32) / 256, 256, 0, stream>>>(W_ih, W_hh, wg);
  pack_int<<<(161 * 16 * 32 + 255) / 256, 256, 0, stream>>>(W_int, wi);
  pack_out<<<(32 * 48 * 32) / 256, 256, 0, stream>>>(W_out, wo);
  dnc_main<<<1, 1024, SMEM_BYTES, stream>>>(source, emb, b_lstm, b_int, b_out,
                                            wg, wi, wo, xiW, outp);
}
